// model_46136538693852
// MI455X (gfx1250) — matrix operand order repaired and checked by reading, not compiled, not run on hardware
//
#include <hip/hip_runtime.h>
#include <hip/hip_bf16.h>
#include <hip/hip_fp16.h>
#include <stdint.h>

typedef _Float16 v16h __attribute__((ext_vector_type(16)));
typedef _Float16 v8h  __attribute__((ext_vector_type(8)));
typedef float    v8f  __attribute__((ext_vector_type(8)));

#define N_NODES  65536
#define N_GRAPHS 256
#define PER_G    256
#define N_EDGES  524288

// ---------------- WMMA helpers (gfx1250, wave32) ----------------

__device__ __forceinline__ v8f vzero8() {
  v8f z;
#pragma unroll
  for (int i = 0; i < 8; ++i) z[i] = 0.f;
  return z;
}

__device__ __forceinline__ v8f wmma16(v16h a, v16h b, v8f c) {
  // D = A(16x32 f16) * B(32x16 f16) + C(16x16 f32)
  return __builtin_amdgcn_wmma_f32_16x16x32_f16(false, a, false, b, (short)0, c,
                                                false, false);
}

// A tile 16x32 from row-major f16; base -> (row0, kk); lda (halves) multiple of 8,
// 16B-aligned. Per-lane data is two contiguous 8-half runs -> two b128 loads.
__device__ __forceinline__ v16h load_a(const _Float16* base, int lda, int g, int ln) {
  const _Float16* p = base + (size_t)ln * lda + (g << 4);
  v8h lo = *(const v8h*)p;
  v8h hi = *(const v8h*)(p + 8);
  v16h a;
#pragma unroll
  for (int i = 0; i < 8; ++i) { a[i] = lo[i]; a[8 + i] = hi[i]; }
  return a;
}

// Packed-B tile: one v16h per lane (32B contiguous).
// Packing rule for a 32x16 tile: element (krel, nrel) lives at
//   lane = nrel + 16*(krel>>4), half = krel & 15.

// Pack full weight matrix W (f32, row-major K x NC) into packed LDS tiles
// laid out [kt][nt][lane][16h]; KT = K/32, NT = NC/16.
template <int NC>
__device__ __forceinline__ void pack_w(const float* __restrict__ W, _Float16* dst,
                                       int K, int tid) {
  const int NT = NC >> 4;
  for (int i = tid; i < K * NC; i += 256) {
    int kr = i / NC, nc = i % NC;
    int kt = kr >> 5, krel = kr & 31;
    int lane = (nc & 15) + ((krel >> 4) << 4);
    int h = krel & 15;
    dst[(((kt * NT + (nc >> 4)) << 5) + lane) * 16 + h] = (_Float16)W[i];
  }
}

// Global pre-pack: W (f32, K x N) -> packed f16 tiles [kt][nt][lane][16h],
// N padded up to NT*16 with zeros.
__global__ __launch_bounds__(256) void k_pack(const float* __restrict__ W,
                                              _Float16* __restrict__ dst,
                                              int K, int N, int NT, int total) {
  int i = blockIdx.x * 256 + threadIdx.x;
  if (i >= total) return;
  int NP = NT << 4;
  int kr = i / NP, nc = i % NP;
  float v = (nc < N) ? W[(size_t)kr * N + nc] : 0.f;
  int kt = kr >> 5, krel = kr & 31;
  int lane = (nc & 15) + ((krel >> 4) << 4);
  int h = krel & 15;
  dst[((((size_t)(kt * NT + (nc >> 4))) << 5) + lane) * 16 + h] = (_Float16)v;
}

// Store WMMA C results (4 n-tiles over a 16x64 tile) into packed-A layout with
// bias + relu. Element (m = r+8g, n = nt*16+ln) -> kt = n>>5, krel = n&31,
// lane = m + 16*(krel>>4), half = krel&15.
__device__ __forceinline__ void store_h_packed(_Float16* hh, const v8f acc[4],
                                               const float* bl, int g, int ln) {
#pragma unroll
  for (int nt = 0; nt < 4; ++nt) {
    float bi = bl[nt * 16 + ln];
    int kt = nt >> 1;
    int krel = ((nt & 1) << 4) + ln;
    int lg = krel >> 4;
    int h = krel & 15;
#pragma unroll
    for (int r = 0; r < 8; ++r) {
      int lanew = r + 8 * g + (lg << 4);
      hh[(((kt << 5) + lanew) << 4) + h] = (_Float16)fmaxf(acc[nt][r] + bi, 0.f);
    }
  }
}

// ---------------- Edge feature kernel ----------------

__global__ __launch_bounds__(256) void k_edge_feat(const float* __restrict__ x,
                                                   const int* __restrict__ eidx,
                                                   _Float16* __restrict__ e16,
                                                   float* __restrict__ maskf) {
  int e = blockIdx.x * 256 + threadIdx.x;
  if (e >= N_EDGES) return;
  int s = eidx[2 * e], r = eidx[2 * e + 1];
  const float* xs = x + (size_t)s * 6;
  const float* xr = x + (size_t)r * 6;
  float d0 = xr[0] - xs[0], d1 = xr[1] - xs[1], d2 = xr[2] - xs[2];
  float dd = sqrtf(d0 * d0 + d1 * d1 + d2 * d2);
  float inv = (dd > 0.f) ? 1.f / dd : 0.f;
  v8h ev;
  ev[0] = (_Float16)(xr[3] - xs[3]);
  ev[1] = (_Float16)(xr[4] - xs[4]);
  ev[2] = (_Float16)(xr[5] - xs[5]);
  ev[3] = (_Float16)dd;
  ev[4] = (_Float16)(d0 * inv);
  ev[5] = (_Float16)(d1 * inv);
  ev[6] = (_Float16)(d2 * inv);
  ev[7] = (_Float16)0.f;
  *(v8h*)(e16 + (size_t)e * 8) = ev;
  maskf[e] = (xs[3] <= xr[3]) ? 1.f : 0.f;
}

// ---------------- Fused edge MLP + ECC message + scatter ----------------

__device__ __forceinline__ void hh_layer_p(v16h* hp, const v16h* Wp, const float* bl,
                                           int g, int ln, int lane) {
  v16h a0 = hp[lane];
  v16h a1 = hp[32 + lane];
  v8f acc[4];
#pragma unroll
  for (int nt = 0; nt < 4; ++nt) {
    acc[nt] = vzero8();
    acc[nt] = wmma16(a0, Wp[(nt << 5) + lane], acc[nt]);
    acc[nt] = wmma16(a1, Wp[((4 + nt) << 5) + lane], acc[nt]);
  }
  store_h_packed((_Float16*)hp, acc, bl, g, ln);
}

__global__ __launch_bounds__(256) void k_edge_mlp(
    const _Float16* __restrict__ e16, const float* __restrict__ maskf,
    const int* __restrict__ eidx, const float* __restrict__ x,
    const float* __restrict__ W0, const float* __restrict__ b0,
    const float* __restrict__ W1, const float* __restrict__ b1,
    const float* __restrict__ W2, const float* __restrict__ b2,
    const float* __restrict__ Wf, const float* __restrict__ bf,
    float* __restrict__ out0) {
  __shared__ v16h sW0p[4 * 32];    // 1 kt x 4 nt (7->64 zero-padded to K=32)
  __shared__ v16h sW1p[8 * 32];    // 2 kt x 4 nt
  __shared__ v16h sW2p[8 * 32];
  __shared__ v16h sWfp[48 * 32];   // 2 kt x 24 nt (kfinal 64->384)
  __shared__ float sB[576];        // b0|b1|b2|bf
  __shared__ v16h sHp[8][2 * 32];  // per-wave packed h staging
  __shared__ float sX[8][16 * 8];  // per-wave x[send](6) + mask
  __shared__ int sR[8][16];        // per-wave recv indices

  // stage weights (packed B layout), reused ~64x per block
  for (int i = threadIdx.x; i < 32 * 64; i += 256) {
    int kr = i >> 6, nc = i & 63;
    float v = (kr < 7) ? W0[kr * 64 + nc] : 0.f;
    int lane = (nc & 15) + ((kr >> 4) << 4);
    ((_Float16*)sW0p)[((((nc >> 4) << 5) + lane) << 4) + (kr & 15)] = (_Float16)v;
  }
  pack_w<64>(W1, (_Float16*)sW1p, 64, threadIdx.x);
  pack_w<64>(W2, (_Float16*)sW2p, 64, threadIdx.x);
  pack_w<384>(Wf, (_Float16*)sWfp, 64, threadIdx.x);
  for (int i = threadIdx.x; i < 576; i += 256) {
    float v;
    if (i < 64) v = b0[i];
    else if (i < 128) v = b1[i - 64];
    else if (i < 192) v = b2[i - 128];
    else v = bf[i - 192];
    sB[i] = v;
  }
  __syncthreads();

  const int wave = threadIdx.x >> 5, lane = threadIdx.x & 31;
  const int g = lane >> 4, ln = lane & 15;
  v16h* hp = sHp[wave];
  float* xsw = sX[wave];
  int* rw = sR[wave];
  const int ntiles = N_EDGES / 16;

  for (int t = blockIdx.x * 8 + wave; t < ntiles; t += gridDim.x * 8) {
    int e0 = t << 4;
    if (lane < 16) {
      int e = e0 + lane;
      int s = eidx[2 * e];
      rw[lane] = eidx[2 * e + 1];
#pragma unroll
      for (int f = 0; f < 6; ++f) xsw[lane * 8 + f] = x[(size_t)s * 6 + f];
      xsw[lane * 8 + 6] = maskf[e];
    }
    // ---- layer 0: e (16x8, zero-padded to 16x32) @ W0 ----
    v16h a;
    {
      const v8h ev = *(const v8h*)(e16 + (size_t)(e0 + ln) * 8);
#pragma unroll
      for (int v = 0; v < 8; ++v) {
        int k = (v << 1) + (g << 4);
        bool ok = (k < 8);
        a[2 * v]     = ok ? ev[k & 7] : (_Float16)0.f;
        a[2 * v + 1] = ok ? ev[(k + 1) & 7] : (_Float16)0.f;
      }
    }
    {
      v8f acc[4];
#pragma unroll
      for (int nt = 0; nt < 4; ++nt) {
        acc[nt] = vzero8();
        acc[nt] = wmma16(a, sW0p[(nt << 5) + lane], acc[nt]);
      }
      store_h_packed((_Float16*)hp, acc, sB, g, ln);
    }
    hh_layer_p(hp, sW1p, sB + 64, g, ln, lane);
    hh_layer_p(hp, sW2p, sB + 128, g, ln, lane);

    // ---- kfinal column-block f == kernel[:, f, :]; msg = sum_f xs_f*(h3@Wf_f + bf_f) ----
    v16h a0 = hp[lane];
    v16h a1 = hp[32 + lane];
    v8f macc[4];
#pragma unroll
    for (int nt = 0; nt < 4; ++nt) macc[nt] = vzero8();
#pragma unroll
    for (int f = 0; f < 6; ++f) {
#pragma unroll
      for (int nt = 0; nt < 4; ++nt) {
        int tb = (f << 2) + nt;
        v8f kt = vzero8();
        kt = wmma16(a0, sWfp[(tb << 5) + lane], kt);
        kt = wmma16(a1, sWfp[((24 + tb) << 5) + lane], kt);
        float bi = sB[192 + (tb << 4) + ln];
#pragma unroll
        for (int r = 0; r < 8; ++r) {
          int m = r + 8 * g;
          macc[nt][r] += xsw[m * 8 + f] * (kt[r] + bi);
        }
      }
    }
    // masked scatter into recv accumulator
#pragma unroll
    for (int nt = 0; nt < 4; ++nt) {
#pragma unroll
      for (int r = 0; r < 8; ++r) {
        int m = r + 8 * g;
        float v = macc[nt][r] * xsw[m * 8 + 6];
        atomicAdd(out0 + (size_t)rw[m] * 64 + (nt << 4) + ln, v);
      }
    }
  }
}

// ---------------- root transform + relu ----------------

__global__ __launch_bounds__(256) void k_root(const float* __restrict__ out0,
                                              const float* __restrict__ x,
                                              const float* __restrict__ root,
                                              const float* __restrict__ bias,
                                              _Float16* __restrict__ xh16) {
  int i = blockIdx.x * 256 + threadIdx.x;  // N_NODES*64 threads
  int n = i >> 6, c = i & 63;
  float s = out0[i] + bias[c];
  const float* xp = x + (size_t)n * 6;
#pragma unroll
  for (int f = 0; f < 6; ++f) s += xp[f] * root[f * 64 + c];
  xh16[i] = (_Float16)fmaxf(s, 0.f);
}

// ---------------- Generic WMMA GEMM (B pre-packed in global, no LDS/barriers) ----
// mode 0: raw -> outH                 (y = xh@W1)
// mode 1: relu(+add+bias) -> outH     (GC combine)
// mode 2: leaky(0.1)+batchnorm -> outH (decoder)
// mode 3: +bias -> outF               (final projection, N may be <64)

__global__ __launch_bounds__(256) void k_gemm(
    const _Float16* __restrict__ A, const _Float16* __restrict__ Bp,
    const float* __restrict__ bias, const float* __restrict__ add,
    const float* __restrict__ gmm, const float* __restrict__ bta,
    const float* __restrict__ mean, const float* __restrict__ var,
    float* __restrict__ outF, _Float16* __restrict__ outH,
    int K, int N, int NT, int mode) {
  const int nt0 = blockIdx.x * 4;
  const int lane = threadIdx.x & 31;
  const int g = lane >> 4, ln = lane & 15;
  const int mw = blockIdx.y * 128 + (threadIdx.x >> 5) * 16;
  v8f acc[4];
#pragma unroll
  for (int nt = 0; nt < 4; ++nt) acc[nt] = vzero8();

  const int KT = K >> 5;
  for (int kt = 0; kt < KT; ++kt) {
    if (kt + 1 < KT)  // cover L2 latency on the next k-step's B tiles
      __builtin_prefetch(Bp + ((size_t)((kt + 1) * NT + nt0) << 9) + lane * 16, 0, 3);
    v16h a = load_a(A + (size_t)mw * K + (kt << 5), K, g, ln);
#pragma unroll
    for (int nt = 0; nt < 4; ++nt) {
      if (nt0 + nt < NT) {
        v16h b = *(const v16h*)(Bp + ((((size_t)(kt * NT + nt0 + nt)) << 5) + lane) * 16);
        acc[nt] = wmma16(a, b, acc[nt]);
      }
    }
  }

#pragma unroll
  for (int nt = 0; nt < 4; ++nt) {
    int n = (nt0 + nt) << 4;
    n += ln;
    if (n >= N) continue;
    float bi = bias ? bias[n] : 0.f;
    float ga = 0.f, be = 0.f, mu = 0.f, iv = 0.f;
    if (mode == 2) {
      ga = gmm[n]; be = bta[n]; mu = mean[n];
      iv = rsqrtf(var[n] + 1e-3f);
    }
#pragma unroll
    for (int r = 0; r < 8; ++r) {
      int m = mw + r + 8 * g;
      float v = acc[nt][r] + bi;
      if (add) v += add[(size_t)m * N + n];
      if (mode == 1) v = fmaxf(v, 0.f);
      else if (mode == 2) {
        v = (v > 0.f) ? v : 0.1f * v;
        v = ga * (v - mu) * iv + be;
      }
      if (outF) outF[(size_t)m * N + n] = v;
      if (outH) outH[(size_t)m * N + n] = (_Float16)v;
    }
  }
}

// ---------------- edge scatter: agg[send] += y[recv] * mask ----------------

__global__ __launch_bounds__(256) void k_scatter(const _Float16* __restrict__ y,
                                                 const int* __restrict__ eidx,
                                                 const float* __restrict__ maskf,
                                                 float* __restrict__ agg,
                                                 int fo, int total) {
  int idx = blockIdx.x * 256 + threadIdx.x;
  if (idx >= total) return;
  int cpe = fo >> 3;
  int e = idx / cpe;
  int ch = (idx - e * cpe) << 3;
  if (maskf[e] == 0.f) return;  // ~half the edges skip all atomics
  int s = eidx[2 * e], rv = eidx[2 * e + 1];
  const v8h yv = *(const v8h*)(y + (size_t)rv * fo + ch);
  float* ap = agg + (size_t)s * fo + ch;
#pragma unroll
  for (int j = 0; j < 8; ++j) atomicAdd(ap + j, (float)yv[j]);
}

// ---------------- per-graph pooling: [max | avg | sum] ----------------

__global__ __launch_bounds__(256) void k_pool(const _Float16* __restrict__ xh,
                                              _Float16* __restrict__ z16) {
  int gph = blockIdx.x, c = threadIdx.x;
  const _Float16* base = xh + (size_t)gph * PER_G * 512;
  float mx0 = -3.4e38f, mx1 = -3.4e38f, s0 = 0.f, s1 = 0.f;
  for (int i = 0; i < PER_G; ++i) {
    float v0 = (float)base[(size_t)i * 512 + c];
    float v1 = (float)base[(size_t)i * 512 + c + 256];
    mx0 = fmaxf(mx0, v0); s0 += v0;
    mx1 = fmaxf(mx1, v1); s1 += v1;
  }
  _Float16* zr = z16 + (size_t)gph * 1536;
  zr[c] = (_Float16)mx0;            zr[c + 256] = (_Float16)mx1;
  zr[512 + c] = (_Float16)(s0 * (1.f / 256.f));
  zr[512 + c + 256] = (_Float16)(s1 * (1.f / 256.f));
  zr[1024 + c] = (_Float16)s0;      zr[1024 + c + 256] = (_Float16)s1;
}

// ---------------- workspace layout ----------------

constexpr size_t OFF_E16  = 0;
constexpr size_t OFF_MASK = OFF_E16  + (size_t)N_EDGES * 8 * 2;
constexpr size_t OFF_OUT0 = OFF_MASK + (size_t)N_EDGES * 4;
constexpr size_t OFF_XHA  = OFF_OUT0 + (size_t)N_NODES * 64 * 4;
constexpr size_t OFF_XHB  = OFF_XHA  + (size_t)N_NODES * 512 * 2;
constexpr size_t OFF_Y16  = OFF_XHB  + (size_t)N_NODES * 512 * 2;
constexpr size_t OFF_AGG  = OFF_Y16  + (size_t)N_NODES * 512 * 2;
constexpr size_t OFF_Z16  = OFF_AGG  + (size_t)N_NODES * 512 * 4;
constexpr size_t OFF_DA   = OFF_Z16  + (size_t)N_GRAPHS * 1536 * 2;
constexpr size_t OFF_DB   = OFF_DA   + (size_t)N_GRAPHS * 1024 * 2;
constexpr size_t OFF_WP   = OFF_DB   + (size_t)N_GRAPHS * 1024 * 2;  // packed weights

extern "C" void kernel_launch(void* const* d_in, const int* in_sizes, int n_in,
                              void* d_out, int out_size, void* d_ws, size_t ws_size,
                              hipStream_t stream) {
  (void)in_sizes; (void)n_in; (void)out_size; (void)ws_size;

  const float* x    = (const float*)d_in[0];
  const int*   eidx = (const int*)d_in[1];
  // d_in[2] seg_ids (contiguous by construction), d_in[3] n_graphs: structure fixed.
  int p = 4;
  // params flattened: bn, decode, ecc, gcs, out (sorted keys; nested dicts sorted)
  const float *bnBeta[5], *bnGamma[5], *bnMean[5], *bnVar[5];
  for (int j = 0; j < 5; ++j) {
    bnBeta[j]  = (const float*)d_in[p++];
    bnGamma[j] = (const float*)d_in[p++];
    bnMean[j]  = (const float*)d_in[p++];
    bnVar[j]   = (const float*)d_in[p++];
  }
  const float *decW[5], *decB[5];
  for (int j = 0; j < 5; ++j) { decW[j] = (const float*)d_in[p++]; decB[j] = (const float*)d_in[p++]; }
  const float* eccBias = (const float*)d_in[p++];
  const float* kfW = (const float*)d_in[p++];
  const float* kfB = (const float*)d_in[p++];
  const float *knW[3], *knB[3];
  for (int j = 0; j < 3; ++j) { knW[j] = (const float*)d_in[p++]; knB[j] = (const float*)d_in[p++]; }
  const float* root = (const float*)d_in[p++];
  const float *gW1[4], *gW2[4], *gB[4];
  for (int j = 0; j < 4; ++j) {
    gW1[j] = (const float*)d_in[p++];
    gW2[j] = (const float*)d_in[p++];
    gB[j]  = (const float*)d_in[p++];
  }
  const float* outW = (const float*)d_in[p++];
  const float* outB = (const float*)d_in[p++];

  uint8_t* ws = (uint8_t*)d_ws;
  _Float16* e16   = (_Float16*)(ws + OFF_E16);
  float*    maskf = (float*)   (ws + OFF_MASK);
  float*    out0  = (float*)   (ws + OFF_OUT0);
  _Float16* xhA   = (_Float16*)(ws + OFF_XHA);
  _Float16* xhB   = (_Float16*)(ws + OFF_XHB);
  _Float16* y16   = (_Float16*)(ws + OFF_Y16);
  float*    agg   = (float*)   (ws + OFF_AGG);
  _Float16* z16   = (_Float16*)(ws + OFF_Z16);
  _Float16* dA    = (_Float16*)(ws + OFF_DA);
  _Float16* dB    = (_Float16*)(ws + OFF_DB);

  // 0) pre-pack all GEMM weights into WMMA-B layout (f16, zero-padded N)
  size_t wpOff = OFF_WP;
  auto packW = [&](const float* W, int K, int N) -> const _Float16* {
    int NT = (N + 15) >> 4;
    size_t halves = (size_t)K * NT * 16;
    _Float16* dst = (_Float16*)(ws + wpOff);
    wpOff += halves * 2;
    int total = (int)halves;
    k_pack<<<(total + 255) / 256, 256, 0, stream>>>(W, dst, K, N, NT, total);
    return dst;
  };
  const int fis[4] = {64, 64, 128, 256};
  const int fos[4] = {64, 128, 256, 512};
  const _Float16 *gW1p[4], *gW2p[4];
  for (int l = 0; l < 4; ++l) {
    gW1p[l] = packW(gW1[l], fis[l], fos[l]);
    gW2p[l] = packW(gW2[l], fis[l], fos[l]);
  }
  const int dn[5] = {1024, 512, 256, 128, 128};
  const _Float16* decWp[5];
  int Kc = 1536;
  for (int j = 0; j < 5; ++j) { decWp[j] = packW(decW[j], Kc, dn[j]); Kc = dn[j]; }
  const _Float16* outWp = packW(outW, 128, 7);

  // 1) edge features + mask
  k_edge_feat<<<N_EDGES / 256, 256, 0, stream>>>(x, eidx, e16, maskf);

  // 2) fused edge MLP -> kernel -> message -> segment_sum(recv)
  hipMemsetAsync(out0, 0, (size_t)N_NODES * 64 * 4, stream);
  k_edge_mlp<<<512, 256, 0, stream>>>(e16, maskf, eidx, x,
                                      knW[0], knB[0], knW[1], knB[1], knW[2], knB[2],
                                      kfW, kfB, out0);

  // 3) xh = relu(out0 + x@root + bias)
  k_root<<<N_NODES * 64 / 256, 256, 0, stream>>>(out0, x, root, eccBias, xhA);

  // 4) graph-conv layers
  _Float16* cur = xhA;
  _Float16* nxt = xhB;
  for (int l = 0; l < 4; ++l) {
    int fi = fis[l], fo = fos[l];
    hipMemsetAsync(agg, 0, (size_t)N_NODES * fo * 4, stream);
    k_gemm<<<dim3(fo / 64, N_NODES / 128), 256, 0, stream>>>(
        cur, gW1p[l], nullptr, nullptr, nullptr, nullptr, nullptr, nullptr,
        nullptr, y16, fi, fo, fo / 16, 0);
    int total = N_EDGES * (fo / 8);
    k_scatter<<<total / 256, 256, 0, stream>>>(y16, eidx, maskf, agg, fo, total);
    k_gemm<<<dim3(fo / 64, N_NODES / 128), 256, 0, stream>>>(
        cur, gW2p[l], gB[l], agg, nullptr, nullptr, nullptr, nullptr,
        nullptr, nxt, fi, fo, fo / 16, 1);
    _Float16* t = cur; cur = nxt; nxt = t;
  }
  // final xh in `cur` (N x 512)

  // 5) pooling -> z (256 x 1536)
  k_pool<<<N_GRAPHS, 256, 0, stream>>>(cur, z16);

  // 6) decoder: 5x (dense + leaky_relu + batchnorm)
  const _Float16* din = z16;
  Kc = 1536;
  for (int j = 0; j < 5; ++j) {
    _Float16* o = (j & 1) ? dB : dA;
    k_gemm<<<dim3(dn[j] / 64, N_GRAPHS / 128), 256, 0, stream>>>(
        din, decWp[j], decB[j], nullptr, bnGamma[j], bnBeta[j], bnMean[j], bnVar[j],
        nullptr, o, Kc, dn[j], dn[j] / 16, 2);
    din = o;
    Kc = dn[j];
  }

  // 7) final projection 128 -> 7 (f32 output)
  k_gemm<<<dim3(1, N_GRAPHS / 128), 256, 0, stream>>>(
      din, outWp, outB, nullptr, nullptr, nullptr, nullptr, nullptr,
      (float*)d_out, nullptr, 128, 7, 1, 3);
}
